// cape_64733747085778
// MI455X (gfx1250) — matrix operand order repaired and checked by reading, not compiled, not run on hardware
//
#include <hip/hip_runtime.h>
#include <hip/hip_bf16.h>
#include <math.h>

// ---------------------------------------------------------------------------
// Problem constants (from reference)
// ---------------------------------------------------------------------------
#define BB   8
#define WW   20
#define MM   512
#define FF   16
#define HH   128
#define HALF 64
#define KK   10
#define G3   384           // 3*H
#define NSEQ (BB*MM)       // 4096

typedef __attribute__((ext_vector_type(16))) _Float16 v16h;
typedef __attribute__((ext_vector_type(8)))  float    v8f;

union F16x16 { v16h v; _Float16 h[16]; unsigned u[8]; };

#define WMMA_F16(a, b, c) \
  __builtin_amdgcn_wmma_f32_16x16x32_f16(false, (a), false, (b), (short)0, (c), false, false)

// ---------------------------------------------------------------------------
// WMMA fragment gathers (wave32, 16x16x32 f16, per CDNA5 ISA §7.12.2)
// A (16Mx32K): lane L holds row m=L&15; lanes<16 carry K∈{0..7,16..23},
// lanes>=16 carry K∈{8..15,24..31}; element e -> k=(e/8)*16 + (L>>4)*8 + (e%8)
// ---------------------------------------------------------------------------
__device__ inline v16h load_a_frag(const _Float16* smA, int lda) {
  const int lane = threadIdx.x & 31;
  const int m = lane & 15, hi = lane >> 4;
  const _Float16* rp = smA + m * lda + hi * 8;
  F16x16 a;
#pragma unroll
  for (int p = 0; p < 4; ++p) a.u[p]     = *(const unsigned*)(rp + 2 * p);       // k = hi*8 + 0..7
#pragma unroll
  for (int p = 0; p < 4; ++p) a.u[4 + p] = *(const unsigned*)(rp + 16 + 2 * p);  // k = 16 + hi*8 + 0..7
  return a.v;
}

// B (32Kx16N): lane L holds column n=L&15; element e -> k=(e/8)*16 + (L>>4)*8 + (e%8).
__device__ inline v16h load_b_frag(const _Float16* smB, int ldb) {
  const int lane = threadIdx.x & 31;
  const int n = lane & 15, hi = lane >> 4;
  const _Float16* cp = smB + n;
  F16x16 b;
#pragma unroll
  for (int e = 0; e < 8; ++e) {
    b.h[e]     = cp[(hi * 8 + e) * ldb];
    b.h[8 + e] = cp[(16 + hi * 8 + e) * ldb];
  }
  return b.v;
}
// C/D (16x16 f32): VGPR v, lanes<16 -> (m=v, n=lane); lanes>=16 -> (m=v+8, n=lane-16)

__device__ inline float sigmf(float x) { return 1.f / (1.f + __expf(-x)); }

// ---------------------------------------------------------------------------
// Kernel 1: GRU over W=20 steps. 256 WGs x 256 thr (8 waves). WG owns 16 seqs,
// wave w owns hidden cols [16w,16w+16). All weights LDS-resident (f16,
// transposed to B-layout [k][n]) so WMMA operands stream via ds_load_b128 —
// no scratch spills. LDS: 24KB (WihT pad) + 96KB (WhhT) + 4KB (h) + 1KB (x)
// = 125KB of the WGP's 320KB.
// ---------------------------------------------------------------------------
__global__ __launch_bounds__(256) void gru_kernel(
    const float* __restrict__ X,
    const float* __restrict__ Wih, const float* __restrict__ Whh,
    const float* __restrict__ bih, const float* __restrict__ bhh,
    float* __restrict__ last_hid) {
  __shared__ _Float16 smem[32 * G3 + HH * G3 + 16 * HH + 16 * 32];
  _Float16* sWih = smem;                 // [k=0..31][n=0..383], rows >=16 zero
  _Float16* sWhh = sWih + 32 * G3;       // [k=0..127][n=0..383]
  _Float16* sH   = sWhh + HH * G3;       // 16 x 128
  _Float16* sX   = sH + 16 * HH;         // 16 x 32 (K padded 16->32)

  const int tid  = threadIdx.x;
  const int wave = tid >> 5, lane = tid & 31;
  const int seq0 = blockIdx.x * 16;
  const int b    = seq0 / MM, m0 = seq0 % MM;
  const int j0   = wave * 16;
  const int cl   = lane & 15, hi = lane >> 4;

  // Stage transposed weights into LDS: sW[k][n] = W[n][k].
  for (int i = tid; i < 32 * G3; i += 256) {
    const int k = i / G3, n = i - k * G3;
    sWih[i] = (k < FF) ? (_Float16)Wih[(size_t)n * FF + k] : (_Float16)0.f;
  }
  for (int i = tid; i < HH * G3; i += 256) {
    const int k = i / G3, n = i - k * G3;
    sWhh[i] = (_Float16)Whh[(size_t)n * HH + k];
  }
  for (int i = tid; i < 16 * HH; i += 256) sH[i] = (_Float16)0.f;

  const float bih_r = bih[0 * HH + j0 + cl], bih_z = bih[1 * HH + j0 + cl], bih_n = bih[2 * HH + j0 + cl];
  const float bhh_r = bhh[0 * HH + j0 + cl], bhh_z = bhh[1 * HH + j0 + cl], bhh_n = bhh[2 * HH + j0 + cl];

  float hreg[8];
#pragma unroll
  for (int v = 0; v < 8; ++v) hreg[v] = 0.f;
  __syncthreads();

  for (int t = 0; t < WW; ++t) {
    // stage x_t (16 seqs x F, padded to K=32)
    for (int i = tid; i < 16 * 32; i += 256) {
      const int m = i >> 5, k = i & 31;
      sX[i] = (k < FF)
                ? (_Float16)X[(((size_t)b * WW + t) * MM + (m0 + m)) * FF + k]
                : (_Float16)0.f;
    }
    __syncthreads();

    v8f gxr, gxz, gxn, ghr, ghz, ghn;
#pragma unroll
    for (int v = 0; v < 8; ++v) {
      gxr[v] = bih_r; gxz[v] = bih_z; gxn[v] = bih_n;
      ghr[v] = bhh_r; ghz[v] = bhh_z; ghn[v] = bhh_n;
    }

    const v16h aX = load_a_frag(sX, 32);
    gxr = WMMA_F16(aX, load_b_frag(sWih + 0 * HH + j0, G3), gxr);
    gxz = WMMA_F16(aX, load_b_frag(sWih + 1 * HH + j0, G3), gxz);
    gxn = WMMA_F16(aX, load_b_frag(sWih + 2 * HH + j0, G3), gxn);
#pragma unroll
    for (int kc = 0; kc < 4; ++kc) {
      const v16h aH = load_a_frag(sH + kc * 32, HH);
      const _Float16* wrow = sWhh + (size_t)(kc * 32) * G3 + j0;
      ghr = WMMA_F16(aH, load_b_frag(wrow + 0 * HH, G3), ghr);
      ghz = WMMA_F16(aH, load_b_frag(wrow + 1 * HH, G3), ghz);
      ghn = WMMA_F16(aH, load_b_frag(wrow + 2 * HH, G3), ghn);
    }

#pragma unroll
    for (int v = 0; v < 8; ++v) {
      const float r = sigmf(gxr[v] + ghr[v]);
      const float z = sigmf(gxz[v] + ghz[v]);
      const float n = tanhf(gxn[v] + r * ghn[v]);
      hreg[v] = (1.f - z) * n + z * hreg[v];
    }
    __syncthreads();                      // all waves done reading sH for step t
#pragma unroll
    for (int v = 0; v < 8; ++v)
      sH[(hi * 8 + v) * HH + j0 + cl] = (_Float16)hreg[v];
    __syncthreads();
  }

#pragma unroll
  for (int v = 0; v < 8; ++v)
    last_hid[((size_t)(seq0 + hi * 8 + v)) * HH + j0 + cl] = hreg[v];
}

// ---------------------------------------------------------------------------
// Kernel 2: P1 = LH @ W1^T, P2 = LH @ W2^T   (4096x128 @ 128x64, WMMA)
// 256 WGs x 128 thr; wave w -> output cols [16w,16w+16). W1/W2 LDS-resident.
// ---------------------------------------------------------------------------
__global__ __launch_bounds__(128) void p12_kernel(
    const float* __restrict__ lh,
    const float* __restrict__ W1, const float* __restrict__ W2,
    float* __restrict__ P1, float* __restrict__ P2) {
  __shared__ _Float16 sLH[16 * HH];
  __shared__ _Float16 sW1[HH * HALF];   // [k][n] = W1[n][k]
  __shared__ _Float16 sW2[HH * HALF];
  const int tid = threadIdx.x, wave = tid >> 5, lane = tid & 31;
  const int seq0 = blockIdx.x * 16;
  for (int i = tid; i < 16 * HH; i += 128) sLH[i] = (_Float16)lh[(size_t)seq0 * HH + i];
  for (int i = tid; i < HH * HALF; i += 128) {
    const int k = i >> 6, n = i & (HALF - 1);
    sW1[i] = (_Float16)W1[(size_t)n * HH + k];
    sW2[i] = (_Float16)W2[(size_t)n * HH + k];
  }
  __syncthreads();

  const int n0 = wave * 16;
  v8f a1 = {}, a2 = {};
#pragma unroll
  for (int kc = 0; kc < 4; ++kc) {
    const v16h af = load_a_frag(sLH + kc * 32, HH);
    a1 = WMMA_F16(af, load_b_frag(sW1 + (size_t)(kc * 32) * HALF + n0, HALF), a1);
    a2 = WMMA_F16(af, load_b_frag(sW2 + (size_t)(kc * 32) * HALF + n0, HALF), a2);
  }
  const int cl = lane & 15, hi = lane >> 4;
#pragma unroll
  for (int v = 0; v < 8; ++v) {
    const int row = seq0 + hi * 8 + v;
    P1[(size_t)row * HALF + n0 + cl] = a1[v];
    P2[(size_t)row * HALF + n0 + cl] = a2[v];
  }
}

// ---------------------------------------------------------------------------
// Kernel 3: a_mx[b,i,j] = sum_k elu(P1[b,j,k]+P2[b,i,k]+b1[k])*V[k] + bv
// one block per (b,i)
// ---------------------------------------------------------------------------
__global__ void amx_kernel(const float* __restrict__ P1, const float* __restrict__ P2,
                           const float* __restrict__ b1, const float* __restrict__ V,
                           const float* __restrict__ bv, float* __restrict__ amx) {
  const int bi = blockIdx.x;                 // b*512 + i
  const int bb = bi >> 9;
  __shared__ float sp2[HALF], sb1[HALF], sV[HALF];
  if (threadIdx.x < HALF) {
    sp2[threadIdx.x] = P2[(size_t)bi * HALF + threadIdx.x];
    sb1[threadIdx.x] = b1[threadIdx.x];
    sV[threadIdx.x]  = V[threadIdx.x];
  }
  __syncthreads();
  const float bvv = bv[0];
  for (int j = threadIdx.x; j < MM; j += blockDim.x) {
    const float4* p14 = (const float4*)(P1 + ((size_t)bb * MM + j) * HALF);
    float s = 0.f;
#pragma unroll 4
    for (int q = 0; q < HALF / 4; ++q) {
      const float4 p = p14[q];
      const int k = q * 4;
      float pre, e;
      pre = p.x + sp2[k + 0] + sb1[k + 0]; e = pre > 0.f ? pre : (__expf(pre) - 1.f); s += e * sV[k + 0];
      pre = p.y + sp2[k + 1] + sb1[k + 1]; e = pre > 0.f ? pre : (__expf(pre) - 1.f); s += e * sV[k + 1];
      pre = p.z + sp2[k + 2] + sb1[k + 2]; e = pre > 0.f ? pre : (__expf(pre) - 1.f); s += e * sV[k + 2];
      pre = p.w + sp2[k + 3] + sb1[k + 3]; e = pre > 0.f ? pre : (__expf(pre) - 1.f); s += e * sV[k + 3];
    }
    amx[(size_t)bi * MM + j] = s + bvv;
  }
}

// ---------------------------------------------------------------------------
// Kernel 4: L2-normalize a_mx over dim i (column-wise per (b,j)), in place.
// ---------------------------------------------------------------------------
__global__ void norm_kernel(float* __restrict__ amx) {
  const int idx = blockIdx.x * blockDim.x + threadIdx.x;  // b*512 + j
  if (idx >= BB * MM) return;
  const int bb = idx >> 9, j = idx & (MM - 1);
  float* base = amx + (size_t)bb * MM * MM + j;
  float s = 0.f;
  for (int i = 0; i < MM; ++i) { const float v = base[(size_t)i * MM]; s += v * v; }
  const float inv = 1.f / fmaxf(sqrtf(s), 1e-12f);
  for (int i = 0; i < MM; ++i) base[(size_t)i * MM] *= inv;
}

// ---------------------------------------------------------------------------
// Kernel 5: per-node conv1d branch -> r_l (4096 x 30), relu'd.
// feature index = ch*3 + pos; pos0 = full-width conv, pos1/2 = dilated(2).
// ---------------------------------------------------------------------------
__global__ void conv_kernel(const float* __restrict__ X,
                            const float* __restrict__ cw, const float* __restrict__ cb,
                            const float* __restrict__ clw, const float* __restrict__ clb,
                            float* __restrict__ rl) {
  const int idx = blockIdx.x * blockDim.x + threadIdx.x;
  if (idx >= NSEQ * 3 * KK) return;
  const int bm = idx / (3 * KK), fe = idx % (3 * KK);
  const int ch = fe / 3, pos = fe % 3;
  const int bb = bm >> 9, m = bm & (MM - 1);
  float s;
  if (pos == 0) {
    s = cb[ch];
    for (int w = 0; w < WW; ++w)
#pragma unroll
      for (int f = 0; f < FF; ++f)
        s += X[(((size_t)bb * WW + w) * MM + m) * FF + f] * cw[(ch * FF + f) * WW + w];
  } else {
    const int p = pos - 1;
    s = clb[ch];
    for (int kk = 0; kk < 10; ++kk)
#pragma unroll
      for (int f = 0; f < FF; ++f)
        s += X[(((size_t)bb * WW + (p + 2 * kk)) * MM + m) * FF + f] * clw[(ch * FF + f) * 10 + kk];
  }
  rl[(size_t)bm * (3 * KK) + fe] = fmaxf(s, 0.f);
}

// ---------------------------------------------------------------------------
// Kernel 6: c = sigmoid(a_mx @ Wb + wb); A = adj*c + a_mx*(1-c)  -> f16 A.
// grid (16,16,8), block 128 (4 waves), 32x32 tile/WG, WMMA over K=512.
// ---------------------------------------------------------------------------
__global__ __launch_bounds__(128) void adjgate_kernel(
    const float* __restrict__ amx, const float* __restrict__ Wb,
    const float* __restrict__ wb, const float* __restrict__ adj,
    _Float16* __restrict__ Ah) {
  __shared__ _Float16 sA[32 * 32];
  __shared__ _Float16 sB[32 * 32];
  const int bb = blockIdx.z, m0 = blockIdx.x * 32, n0 = blockIdx.y * 32;
  const int tid = threadIdx.x, wave = tid >> 5, lane = tid & 31;
  const float* Ab = amx + (size_t)bb * MM * MM;
  const int wm = (wave >> 1) * 16, wn = (wave & 1) * 16;
  v8f acc = {};
  for (int kc = 0; kc < MM / 32; ++kc) {
    const int k0 = kc * 32;
    for (int i = tid; i < 1024; i += 128) {
      const int r = i >> 5, c = i & 31;
      sA[i] = (_Float16)Ab[(size_t)(m0 + r) * MM + k0 + c];
      sB[i] = (_Float16)Wb[(size_t)(k0 + r) * MM + n0 + c];
    }
    __syncthreads();
    acc = WMMA_F16(load_a_frag(sA + wm * 32, 32), load_b_frag(sB + wn, 32), acc);
    __syncthreads();
  }
  const float wbb = wb[0];
  const int hi = lane >> 4, cl = lane & 15;
#pragma unroll
  for (int v = 0; v < 8; ++v) {
    const int r = m0 + wm + hi * 8 + v;
    const int c = n0 + wn + cl;
    const float g  = sigmf(acc[v] + wbb);
    const float am = Ab[(size_t)r * MM + c];
    const float a  = adj[(size_t)r * MM + c] * g + am * (1.f - g);
    Ah[((size_t)bb * MM + r) * MM + c] = (_Float16)a;
  }
}

// ---------------------------------------------------------------------------
// Kernel 7: X1 = r_l @ gc1_W  (4096x30 @ 30x128)  — small, VALU.
// ---------------------------------------------------------------------------
__global__ void x1_kernel(const float* __restrict__ rl, const float* __restrict__ gc1W,
                          float* __restrict__ X1) {
  const int idx = blockIdx.x * blockDim.x + threadIdx.x;  // bm*128 + h
  if (idx >= NSEQ * HH) return;
  const int bm = idx >> 7, h = idx & (HH - 1);
  float s = 0.f;
#pragma unroll
  for (int q = 0; q < 3 * KK; ++q) s += rl[(size_t)bm * (3 * KK) + q] * gc1W[q * HH + h];
  X1[idx] = s;
}

// ---------------------------------------------------------------------------
// Kernel 8: h1 = relu(A @ X1 + gc1_b)  per b: 512x512 @ 512x128 (WMMA).
// grid (16,4,8), block 128.
// ---------------------------------------------------------------------------
__global__ __launch_bounds__(128) void spmm1_kernel(
    const _Float16* __restrict__ Ah, const float* __restrict__ X1,
    const float* __restrict__ gc1b, float* __restrict__ h1) {
  __shared__ _Float16 sA[32 * 32];
  __shared__ _Float16 sB[32 * 32];
  const int bb = blockIdx.z, m0 = blockIdx.x * 32, n0 = blockIdx.y * 32;
  const int tid = threadIdx.x, wave = tid >> 5, lane = tid & 31;
  const _Float16* Ab = Ah + (size_t)bb * MM * MM;
  const int wm = (wave >> 1) * 16, wn = (wave & 1) * 16;
  v8f acc = {};
  for (int kc = 0; kc < MM / 32; ++kc) {
    const int k0 = kc * 32;
    for (int i = tid; i < 1024; i += 128) {
      const int r = i >> 5, c = i & 31;
      sA[i] = Ab[(size_t)(m0 + r) * MM + k0 + c];
      sB[i] = (_Float16)X1[((size_t)bb * MM + k0 + r) * HH + n0 + c];
    }
    __syncthreads();
    acc = WMMA_F16(load_a_frag(sA + wm * 32, 32), load_b_frag(sB + wn, 32), acc);
    __syncthreads();
  }
  const int hi = lane >> 4, cl = lane & 15;
#pragma unroll
  for (int v = 0; v < 8; ++v) {
    const int r = m0 + wm + hi * 8 + v;
    const int c = n0 + wn + cl;
    h1[((size_t)bb * MM + r) * HH + c] = fmaxf(acc[v] + gc1b[c], 0.f);
  }
}

// ---------------------------------------------------------------------------
// Kernel 9: X2 = h1 @ gc2_W  (4096x128 @ 128x10) — small, VALU.
// ---------------------------------------------------------------------------
__global__ void x2_kernel(const float* __restrict__ h1, const float* __restrict__ gc2W,
                          float* __restrict__ X2) {
  const int idx = blockIdx.x * blockDim.x + threadIdx.x;
  if (idx >= NSEQ * KK) return;
  const int bm = idx / KK, ch = idx % KK;
  float s = 0.f;
#pragma unroll 8
  for (int h = 0; h < HH; ++h) s += h1[(size_t)bm * HH + h] * gc2W[h * KK + ch];
  X2[idx] = s;
}

// ---------------------------------------------------------------------------
// Kernel 10: out_spatial = relu(A @ X2 + gc2_b)  per b: 512x512 @ 512x10 (WMMA,
// N padded to 16). grid (16,1,8), block 64 (2 waves, 32x16 tile/WG).
// ---------------------------------------------------------------------------
__global__ __launch_bounds__(64) void spmm2_kernel(
    const _Float16* __restrict__ Ah, const float* __restrict__ X2,
    const float* __restrict__ gc2b, float* __restrict__ osp) {
  __shared__ _Float16 sA[32 * 32];
  __shared__ _Float16 sB[32 * 16];
  const int bb = blockIdx.z, m0 = blockIdx.x * 32;
  const int tid = threadIdx.x, wave = tid >> 5, lane = tid & 31;
  const _Float16* Ab = Ah + (size_t)bb * MM * MM;
  v8f acc = {};
  for (int kc = 0; kc < MM / 32; ++kc) {
    const int k0 = kc * 32;
    for (int i = tid; i < 1024; i += 64) {
      const int r = i >> 5, c = i & 31;
      sA[i] = Ab[(size_t)(m0 + r) * MM + k0 + c];
    }
    for (int i = tid; i < 512; i += 64) {
      const int r = i >> 4, c = i & 15;
      sB[i] = (c < KK) ? (_Float16)X2[((size_t)bb * MM + k0 + r) * KK + c] : (_Float16)0.f;
    }
    __syncthreads();
    acc = WMMA_F16(load_a_frag(sA + wave * 16 * 32, 32), load_b_frag(sB, 16), acc);
    __syncthreads();
  }
  const int hi = lane >> 4, cl = lane & 15;
#pragma unroll
  for (int v = 0; v < 8; ++v) {
    const int r = m0 + wave * 16 + hi * 8 + v;
    if (cl < KK) osp[((size_t)bb * MM + r) * KK + cl] = fmaxf(acc[v] + gc2b[cl], 0.f);
  }
}

// ---------------------------------------------------------------------------
// Kernel 11: readout y = [osp, lh] @ out_W^T + out_b; sigmoid -> d_out[1..];
// per-block BCE partial sums -> part[]. Kernel 12 reduces to d_out[0].
// ---------------------------------------------------------------------------
__global__ void readout_kernel(const float* __restrict__ osp, const float* __restrict__ lh,
                               const float* __restrict__ Y, const float* __restrict__ outW,
                               const float* __restrict__ outb,
                               float* __restrict__ dout, float* __restrict__ part) {
  __shared__ float red[256];
  const int bm = blockIdx.x * 256 + threadIdx.x;
  float s = outb[0];
#pragma unroll
  for (int k = 0; k < KK; ++k) s += osp[(size_t)bm * KK + k] * outW[k];
#pragma unroll 8
  for (int h = 0; h < HH; ++h) s += lh[(size_t)bm * HH + h] * outW[KK + h];
  dout[1 + bm] = sigmf(s);
  // softplus(s) - Y*s, numerically stable
  const float li = fmaxf(s, 0.f) + log1pf(__expf(-fabsf(s))) - Y[bm] * s;
  red[threadIdx.x] = li;
  __syncthreads();
  for (int off = 128; off > 0; off >>= 1) {
    if (threadIdx.x < off) red[threadIdx.x] += red[threadIdx.x + off];
    __syncthreads();
  }
  if (threadIdx.x == 0) part[blockIdx.x] = red[0];
}

__global__ void loss_kernel(const float* __restrict__ part, float* __restrict__ dout) {
  if (threadIdx.x == 0) {
    float s = 0.f;
    for (int i = 0; i < 16; ++i) s += part[i];
    dout[0] = s / (float)NSEQ;
  }
}

// ---------------------------------------------------------------------------
// Workspace layout (float offsets)
// ---------------------------------------------------------------------------
#define OFF_LH   ((size_t)0)                       // 4096*128
#define OFF_P1   (OFF_LH   + (size_t)NSEQ * HH)    // 4096*64
#define OFF_P2   (OFF_P1   + (size_t)NSEQ * HALF)
#define OFF_AMX  (OFF_P2   + (size_t)NSEQ * HALF)  // 8*512*512
#define OFF_RL   (OFF_AMX  + (size_t)BB * MM * MM) // 4096*30
#define OFF_X1   (OFF_RL   + (size_t)NSEQ * 3 * KK)
#define OFF_H1   (OFF_X1   + (size_t)NSEQ * HH)
#define OFF_X2   (OFF_H1   + (size_t)NSEQ * HH)
#define OFF_OSP  (OFF_X2   + (size_t)NSEQ * KK)
#define OFF_PART (OFF_OSP  + (size_t)NSEQ * KK)
#define OFF_AH   (OFF_PART + 16)                   // f16 region, 8*512*512 halves

extern "C" void kernel_launch(void* const* d_in, const int* in_sizes, int n_in,
                              void* d_out, int out_size, void* d_ws, size_t ws_size,
                              hipStream_t stream) {
  const float* X     = (const float*)d_in[0];
  const float* Y     = (const float*)d_in[1];
  const float* adj   = (const float*)d_in[2];
  const float* V     = (const float*)d_in[3];
  const float* bv    = (const float*)d_in[4];
  const float* W1    = (const float*)d_in[5];
  const float* b1    = (const float*)d_in[6];
  const float* W2    = (const float*)d_in[7];
  const float* Wb    = (const float*)d_in[8];
  const float* wb    = (const float*)d_in[9];
  const float* cw    = (const float*)d_in[10];
  const float* cb    = (const float*)d_in[11];
  const float* clw   = (const float*)d_in[12];
  const float* clb   = (const float*)d_in[13];
  const float* gWih  = (const float*)d_in[14];
  const float* gWhh  = (const float*)d_in[15];
  const float* gbih  = (const float*)d_in[16];
  const float* gbhh  = (const float*)d_in[17];
  const float* gc1W  = (const float*)d_in[18];
  const float* gc1b  = (const float*)d_in[19];
  const float* gc2W  = (const float*)d_in[20];
  const float* gc2b  = (const float*)d_in[21];
  const float* outW  = (const float*)d_in[22];
  const float* outb  = (const float*)d_in[23];

  float* ws   = (float*)d_ws;
  float* lh   = ws + OFF_LH;
  float* P1   = ws + OFF_P1;
  float* P2   = ws + OFF_P2;
  float* amx  = ws + OFF_AMX;
  float* rl   = ws + OFF_RL;
  float* X1   = ws + OFF_X1;
  float* h1   = ws + OFF_H1;
  float* X2   = ws + OFF_X2;
  float* osp  = ws + OFF_OSP;
  float* part = ws + OFF_PART;
  _Float16* Ah = (_Float16*)(ws + OFF_AH);
  float* dout = (float*)d_out;

  // 1) GRU last hidden (WMMA)
  gru_kernel<<<NSEQ / 16, 256, 0, stream>>>(X, gWih, gWhh, gbih, gbhh, lh);
  // 2) attention projections (WMMA)
  p12_kernel<<<NSEQ / 16, 128, 0, stream>>>(lh, W1, W2, P1, P2);
  // 3) adaptive adjacency scores
  amx_kernel<<<BB * MM, 256, 0, stream>>>(P1, P2, b1, V, bv, amx);
  // 4) column L2 normalize
  norm_kernel<<<(BB * MM + 255) / 256, 256, 0, stream>>>(amx);
  // 5) conv branch
  conv_kernel<<<(NSEQ * 3 * KK + 255) / 256, 256, 0, stream>>>(X, cw, cb, clw, clb, rl);
  // 6) gated adjacency A (WMMA, 512^3 per batch)
  adjgate_kernel<<<dim3(MM / 32, MM / 32, BB), 128, 0, stream>>>(amx, Wb, wb, adj, Ah);
  // 7) r_l @ gc1_W
  x1_kernel<<<(NSEQ * HH + 255) / 256, 256, 0, stream>>>(rl, gc1W, X1);
  // 8) h1 = relu(A @ X1 + b) (WMMA)
  spmm1_kernel<<<dim3(MM / 32, HH / 32, BB), 128, 0, stream>>>(Ah, X1, gc1b, h1);
  // 9) h1 @ gc2_W
  x2_kernel<<<(NSEQ * KK + 255) / 256, 256, 0, stream>>>(h1, gc2W, X2);
  // 10) out_spatial = relu(A @ X2 + b) (WMMA)
  spmm2_kernel<<<dim3(MM / 32, 1, BB), 64, 0, stream>>>(Ah, X2, gc2b, osp);
  // 11) readout + partial BCE
  readout_kernel<<<NSEQ / 256, 256, 0, stream>>>(osp, lh, Y, outW, outb, dout, part);
  // 12) final loss
  loss_kernel<<<1, 64, 0, stream>>>(part, dout);
}
